// BPGNN_58677843198052
// MI455X (gfx1250) — matrix operand order repaired and checked by reading, not compiled, not run on hardware
//
#include <hip/hip_runtime.h>

typedef __attribute__((ext_vector_type(16))) _Float16 v16h;
typedef __attribute__((ext_vector_type(8)))  float    v8f;

#define DIMK 512
#define C 10
#define CP 16   // classes padded to 16 for WMMA
#define NKB (DIMK / 32)   // 16 k-blocks of 32

// ---------------------------------------------------------------------------
// Kernel A: logb0 = log_softmax(x @ W + b)  via v_wmma_f32_16x16x32_f16
// One wave per 16-row tile; K swept in 16 steps of 32.
// A loads: unconditional b128 (row index clamped; junk rows never read back).
// B: pre-swizzled in LDS to fragment order -> contiguous 32B per lane.
// ---------------------------------------------------------------------------
__global__ __launch_bounds__(256)
void bp_logb0_wmma(const float* __restrict__ x, const float* __restrict__ W,
                   const float* __restrict__ bias, float* __restrict__ logb0,
                   int n_nodes, int n_tiles) {
  __shared__ __align__(32) _Float16 Bfrag[NKB * 32 * 16];  // 16 KB
  __shared__ float tile[8][16 * 17];                       // per-wave logits

  const int tid = threadIdx.x;
  // Stage W (f32, [512 x 10]) into B-fragment layout, cols padded 10->16.
  // Fragment layout: lane L<16 half h = B[K=h][N=L]; lane>=16 -> K=16+h, N=L-16.
  for (int i = tid; i < NKB * 32 * 16; i += 256) {
    const int kb   = i >> 9;
    const int ln   = (i >> 4) & 31;
    const int h    = i & 15;
    const int K    = kb * 32 + (ln >> 4) * 16 + h;
    const int n    = ln & 15;
    Bfrag[i] = (n < C) ? (_Float16)W[K * C + n] : (_Float16)0.0f;
  }
  __syncthreads();

  const int wave   = tid >> 5;
  const int lane   = tid & 31;
  const int tileId = blockIdx.x * 8 + wave;
  const bool active = (tileId < n_tiles);

  const int row = lane & 15;   // M within tile (A layout)
  const int kg  = lane >> 4;   // lane-half selects K sub-range
  const int col = lane & 15;   // N (C/D layout)

  v8f acc = {};
  if (active) {
    const int  grow = tileId * 16 + row;
    const long crow = (grow < n_nodes) ? grow : (long)(n_nodes - 1); // clamp
    const float* xp = x + crow * (long)DIMK;
    for (int kb = 0; kb < NKB; ++kb) {
      // A fragment: halfs 0..15 -> K = kg*16+h
      const float4* p = (const float4*)(xp + kb * 32 + kg * 16);
      const float4* q = (const float4*)(xp + kb * 32 + kg * 16 + 8);
      const float4 a0 = p[0], a1 = p[1], a2 = q[0], a3 = q[1];
      v16h a;
      a[0]  = (_Float16)a0.x; a[1]  = (_Float16)a0.y;
      a[2]  = (_Float16)a0.z; a[3]  = (_Float16)a0.w;
      a[4]  = (_Float16)a1.x; a[5]  = (_Float16)a1.y;
      a[6]  = (_Float16)a1.z; a[7]  = (_Float16)a1.w;
      a[8]  = (_Float16)a2.x; a[9]  = (_Float16)a2.y;
      a[10] = (_Float16)a2.z; a[11] = (_Float16)a2.w;
      a[12] = (_Float16)a3.x; a[13] = (_Float16)a3.y;
      a[14] = (_Float16)a3.z; a[15] = (_Float16)a3.w;
      const v16h bf = *(const v16h*)&Bfrag[(kb * 32 + lane) * 16];
      acc = __builtin_amdgcn_wmma_f32_16x16x32_f16(
          false, a, false, bf, (short)0, acc, false, false);
    }
    float* tl = tile[wave];
#pragma unroll
    for (int r = 0; r < 8; ++r)            // D: elem r -> M = kg*8 + r, N = col
      tl[(kg * 8 + r) * 17 + col] = acc[r];
  }
  __syncthreads();

  if (active && lane < 16) {
    const int node = tileId * 16 + lane;
    if (node < n_nodes) {
      const float* tl = tile[wave] + lane * 17;
      float v[C], m = -3.0e38f;
#pragma unroll
      for (int c = 0; c < C; ++c) { v[c] = tl[c] + bias[c]; m = fmaxf(m, v[c]); }
      float s = 0.0f;
#pragma unroll
      for (int c = 0; c < C; ++c) s += __expf(v[c] - m);
      const float lse = m + __logf(s);
#pragma unroll
      for (int c = 0; c < C; ++c) logb0[node * C + c] = v[c] - lse;
    }
  }
}

// ---------------------------------------------------------------------------
// Kernel B: logH[10][10] from param (tril logsigmoid(10*p), symmetrized, 0 diag)
// ---------------------------------------------------------------------------
__global__ void bp_logH(const float* __restrict__ param, float* __restrict__ logH) {
  int t = threadIdx.x;
  if (t < C * C) {
    int i = t / C, j = t % C;
    float v = 0.0f;
    if (i != j) {
      int r = (i > j) ? i : j, q = (i > j) ? j : i;
      float z = param[r * (r + 1) / 2 + q] * 10.0f;
      v = (z >= 0.0f) ? -log1pf(__expf(-z)) : (z - log1pf(__expf(z)));
    }
    logH[t] = v;
  }
}

// ---------------------------------------------------------------------------
// Generic fill
// ---------------------------------------------------------------------------
__global__ void bp_fill(float* __restrict__ p, float val, long n) {
  long i = (long)blockIdx.x * blockDim.x + threadIdx.x;
  if (i < n) p[i] = val;
}

// ---------------------------------------------------------------------------
// Kernel C: per-edge message update + fused scatter-add into agg
// ---------------------------------------------------------------------------
__global__ __launch_bounds__(256)
void bp_edge(const int* __restrict__ src, const int* __restrict__ dst,
             const int* __restrict__ rv, const float* __restrict__ ew,
             const float* __restrict__ logH, const float* __restrict__ logb,
             const float* __restrict__ msg_in, float* __restrict__ msg_out,
             float* __restrict__ agg, int E) {
  __shared__ float sH[C * C];
  if (threadIdx.x < C * C) sH[threadIdx.x] = logH[threadIdx.x];
  __syncthreads();

  const int e = blockIdx.x * blockDim.x + threadIdx.x;
  if (e >= E) return;

  const int   s = src[e], d = dst[e], r = rv[e];
  const float w = ew[e];

  float t[C];
#pragma unroll
  for (int c = 0; c < C; ++c)
    t[c] = logb[s * C + c] - msg_in[(long)r * C + c];

  float raw[C];
#pragma unroll
  for (int c2 = 0; c2 < C; ++c2) {
    float v[C], m = -3.0e38f;
#pragma unroll
    for (int c1 = 0; c1 < C; ++c1) {
      v[c1] = fmaf(w, sH[c1 * C + c2], t[c1]);
      m = fmaxf(m, v[c1]);
    }
    float ssum = 0.0f;
#pragma unroll
    for (int c1 = 0; c1 < C; ++c1) ssum += __expf(v[c1] - m);
    raw[c2] = m + __logf(ssum);
  }

  float m2 = -3.0e38f;
#pragma unroll
  for (int c = 0; c < C; ++c) m2 = fmaxf(m2, raw[c]);
  float s2 = 0.0f;
#pragma unroll
  for (int c = 0; c < C; ++c) s2 += __expf(raw[c] - m2);
  const float lse = m2 + __logf(s2);

#pragma unroll
  for (int c = 0; c < C; ++c) {
    const float mg = raw[c] - lse;
    msg_out[(long)e * C + c] = mg;
    atomicAdd(&agg[d * C + c], mg);
  }
}

// ---------------------------------------------------------------------------
// Kernel D: log_b = log_normalize(agg + logb0)
// ---------------------------------------------------------------------------
__global__ void bp_node(const float* __restrict__ agg,
                        const float* __restrict__ logb0,
                        float* __restrict__ out, int N) {
  const int n = blockIdx.x * blockDim.x + threadIdx.x;
  if (n >= N) return;
  float v[C], m = -3.0e38f;
#pragma unroll
  for (int c = 0; c < C; ++c) { v[c] = agg[n * C + c] + logb0[n * C + c]; m = fmaxf(m, v[c]); }
  float s = 0.0f;
#pragma unroll
  for (int c = 0; c < C; ++c) s += __expf(v[c] - m);
  const float lse = m + __logf(s);
#pragma unroll
  for (int c = 0; c < C; ++c) out[n * C + c] = v[c] - lse;
}

// ---------------------------------------------------------------------------
extern "C" void kernel_launch(void* const* d_in, const int* in_sizes, int n_in,
                              void* d_out, int out_size, void* d_ws, size_t ws_size,
                              hipStream_t stream) {
  const float* x     = (const float*)d_in[0];
  const int*   ei    = (const int*)  d_in[1];
  const float* ew    = (const float*)d_in[2];
  const int*   rv    = (const int*)  d_in[3];
  const float* W     = (const float*)d_in[4];
  const float* bias  = (const float*)d_in[5];
  const float* param = (const float*)d_in[6];

  const int N = in_sizes[0] / DIMK;
  const int E = in_sizes[2];
  const int* src = ei;
  const int* dst = ei + E;

  // workspace carve-up (floats), 256-element aligned slabs
  const long NC  = (((long)N * C) + 255) & ~255L;
  const long EC  = (((long)E * C) + 255) & ~255L;
  float* logb0 = (float*)d_ws;
  float* logb  = logb0 + NC;
  float* agg   = logb  + NC;
  float* logH  = agg   + NC;
  float* msgA  = logH  + 256;
  float* msgB  = msgA  + EC;

  const int tiles = (N + 15) / 16;

  bp_logH<<<1, 128, 0, stream>>>(param, logH);
  bp_logb0_wmma<<<(tiles + 7) / 8, 256, 0, stream>>>(x, W, bias, logb0, N, tiles);
  bp_fill<<<(int)((((long)E * C) + 255) / 256), 256, 0, stream>>>(
      msgA, -2.302585093f /* -log(10) */, (long)E * C);

  float* msg_in  = msgA;
  float* msg_out = msgB;
  for (int k = 0; k < 5; ++k) {
    bp_fill<<<(int)((((long)N * C) + 255) / 256), 256, 0, stream>>>(
        agg, 0.0f, (long)N * C);
    bp_edge<<<(E + 255) / 256, 256, 0, stream>>>(
        src, dst, rv, ew, logH, (k == 0) ? logb0 : logb,
        msg_in, msg_out, agg, E);
    bp_node<<<(N + 255) / 256, 256, 0, stream>>>(
        agg, logb0, (k == 4) ? (float*)d_out : logb, N);
    float* t = msg_in; msg_in = msg_out; msg_out = t;
  }
}
